// MultiHeadAttention_17781164605968
// MI455X (gfx1250) — compile-verified, matrix operand order checked by reading, not run on hardware
//
#include <hip/hip_runtime.h>
#include <hip/hip_bf16.h>
#include <stdint.h>

#define B_ 2
#define S_ 2048
#define D_ 1024
#define H_ 16
#define DEPTH_ 64

typedef unsigned short u16;
typedef __attribute__((ext_vector_type(16))) __bf16 v16bf;
typedef __attribute__((ext_vector_type(8)))  float  v8f;
typedef __attribute__((ext_vector_type(8)))  u16    u16x8;
typedef __attribute__((ext_vector_type(4)))  unsigned int u32x4;
typedef __attribute__((ext_vector_type(8)))  int    i32x8;
typedef __attribute__((ext_vector_type(4)))  int    i32x4;

union BFrag { v16bf v; u16x8 h[2]; };

#if __has_builtin(__builtin_amdgcn_tensor_load_to_lds)
#define HAVE_TDM 1
#else
#define HAVE_TDM 0
#endif

__device__ __forceinline__ u16 f2b(float f) {
  unsigned u = __float_as_uint(f);
  u += 0x7FFFu + ((u >> 16) & 1u);   // round-to-nearest-even
  return (u16)(u >> 16);
}

__device__ __forceinline__ float rmax16(float v) {
#pragma unroll
  for (int m = 1; m < 16; m <<= 1) v = fmaxf(v, __shfl_xor(v, m, 32));
  return v;
}
__device__ __forceinline__ float rsum16(float v) {
#pragma unroll
  for (int m = 1; m < 16; m <<= 1) v += __shfl_xor(v, m, 32);
  return v;
}

#if HAVE_TDM
// 2D TDM descriptor load: tile (tl1 rows x tl0 elems) of bf16 from global (row
// stride st0 elems) into LDS at byte offset ldsOff.  D# packing per
// CDNA5 ISA.  6-arg builtin form
// (amdgpu-toolchain / therock-10.0 headers).
__device__ __forceinline__ void tdm_issue_2d(unsigned ldsOff, const u16* g,
                                             unsigned td0, unsigned td1,
                                             unsigned long long st0,
                                             unsigned tl0, unsigned tl1) {
  unsigned long long ga = (unsigned long long)(uintptr_t)g;
  u32x4 g0;
  g0[0] = 1u;                                   // count=1, user descriptor
  g0[1] = ldsOff;                               // lds_addr (bytes)
  g0[2] = (unsigned)(ga & 0xFFFFFFFFu);         // global_addr[31:0]
  g0[3] = (unsigned)((ga >> 32) & 0x1FFFFFFu)   // global_addr[56:32]
        | (2u << 30);                           // type=2 ("image")
  i32x8 g1;
  g1[0] = (int)(1u << 16);                      // data_size=1 -> 2 bytes
  g1[1] = (int)((td0 & 0xFFFFu) << 16);         // tensor_dim0[15:0]
  g1[2] = (int)((td0 >> 16) | ((td1 & 0xFFFFu) << 16));   // td0[31:16] | td1[15:0]
  g1[3] = (int)((td1 >> 16) | (tl0 << 16));     // td1[31:16] | tile_dim0
  g1[4] = (int)(tl1 & 0xFFFFu);                 // tile_dim1 (tile_dim2=0)
  g1[5] = (int)(unsigned)(st0 & 0xFFFFFFFFu);   // tensor_dim0_stride[31:0]
  g1[6] = (int)(unsigned)((st0 >> 32) & 0xFFFFu);// tensor_dim0_stride[47:32]
  g1[7] = 0;
  i32x4 z4 = {0, 0, 0, 0};                      // groups 2/3 unused (2D)
  i32x8 z8 = {0, 0, 0, 0, 0, 0, 0, 0};
  __builtin_amdgcn_tensor_load_to_lds(g0, g1, z4, z4, z8, 0);
}
#else
__device__ __forceinline__ void coop_copy_2d(u16* dst, const u16* g,
                                             int rows, int cols, int gstride) {
  const int total = rows * cols / 8;
  for (int i = threadIdx.x; i < total; i += blockDim.x) {
    const int r = (i * 8) / cols, c = (i * 8) % cols;
    *(u16x8*)(dst + r * cols + c) = *(const u16x8*)(g + (size_t)r * gstride + c);
  }
}
#endif

// ---------------- fp32 -> bf16 conversion ----------------
__global__ void cvt_bf16_kernel(const float* __restrict__ s, u16* __restrict__ d, int n) {
  int i = blockIdx.x * blockDim.x + threadIdx.x;
  int stride = gridDim.x * blockDim.x;
  for (; i < n; i += stride) d[i] = f2b(s[i]);
}

// ---------------- Y = X(bf16 MxK) @ W^T(bf16 NxK) + bias ----------------
// mode 0: outB bf16, split heads  (b,h,s,dd)
// mode 1: outB bf16, transposed heads (b,h,dd,s)     [for V]
// mode 2: outF fp32, plain row-major (M x N)
__global__ void gemm_xwt_kernel(const u16* __restrict__ X, const u16* __restrict__ Wt,
                                const float* __restrict__ bias,
                                u16* __restrict__ outB, float* __restrict__ outF,
                                int M, int N, int K, int mode) {
  const int lane  = threadIdx.x & 31;
  const int lm    = lane & 15;
  const int half8 = (lane >> 4) << 3;           // 0 or 8
  const int nq    = N >> 6;                      // N / 64
  const int gw    = blockIdx.x * (blockDim.x >> 5) + (threadIdx.x >> 5);
  if (gw >= (M >> 4) * nq) return;
  const int m0 = (gw / nq) << 4;
  const int n0 = (gw % nq) << 6;

  const u16* xrow = X + (size_t)(m0 + lm) * K;
  const u16* wrow[4];
#pragma unroll
  for (int t = 0; t < 4; ++t) wrow[t] = Wt + (size_t)(n0 + t * 16 + lm) * K;

  v8f acc[4] = {v8f{}, v8f{}, v8f{}, v8f{}};
  for (int k0 = 0; k0 < K; k0 += 32) {
    const int ks = k0 + half8;
    __builtin_prefetch(xrow + ks + 256, 0, 0);
    BFrag a;
    a.h[0] = *(const u16x8*)(xrow + ks);
    a.h[1] = *(const u16x8*)(xrow + ks + 16);
#pragma unroll
    for (int t = 0; t < 4; ++t) {
      BFrag b;
      b.h[0] = *(const u16x8*)(wrow[t] + ks);
      b.h[1] = *(const u16x8*)(wrow[t] + ks + 16);
      acc[t] = __builtin_amdgcn_wmma_f32_16x16x32_bf16(
          false, a.v, false, b.v, (short)0, acc[t], false, false);
    }
  }

#pragma unroll
  for (int t = 0; t < 4; ++t) {
    const int n  = n0 + t * 16 + lm;
    const float bn = bias[n];
#pragma unroll
    for (int r = 0; r < 8; ++r) {
      const int   m   = m0 + r + half8;
      const float val = acc[t][r] + bn;
      if (mode == 2) {
        outF[(size_t)m * N + n] = val;
      } else {
        const u16 bv = f2b(val);
        const int bb = m >> 11, ss = m & (S_ - 1);
        const int hh = n >> 6,  dd = n & 63;
        if (mode == 0)
          outB[(((size_t)bb * H_ + hh) * S_ + ss) * DEPTH_ + dd] = bv;
        else
          outB[(((size_t)bb * H_ + hh) * DEPTH_ + dd) * S_ + ss] = bv;
      }
    }
  }
}

// ---------------- fused attention ----------------
// Qp,Kp: bf16 (B,H,S,64) ; Vt: bf16 (B,H,64,S)
// attnOut: fp32 (B,H,S,S) normalized ; ctxB: bf16 (B,S,D) merged heads
//
// 8 waves/WG all share one (b,h): K/V stream chunked 32 keys at a time into
// double-buffered LDS via TDM (wave 0 issues, s_wait_tensorcnt + s_barrier).
// 3 phases over the K stream: (A) row max, (B) exp-sum, (C) normalized attn
// write + ctx accumulation — per-lane accumulators, one shuffle-reduction per
// phase instead of per tile.
#define NCH   (S_ / 32)      // 64 chunks of 32 keys
#define KBUF_BYTES (32 * 64 * 2)
#define VBUF_BYTES (64 * 32 * 2)

struct SMem {
  u16 kbuf[2][32 * 64];     // offset 0
  u16 vbuf[2][64 * 32];     // offset 8192
  u16 pst[8 * 16 * 32];     // offset 16384
};

__global__ void attn_kernel(const u16* __restrict__ Qp, const u16* __restrict__ Kp,
                            const u16* __restrict__ Vt,
                            float* __restrict__ attnOut, u16* __restrict__ ctxB) {
  __shared__ SMem sm;
  const int lane  = threadIdx.x & 31;
  const int lm    = lane & 15;
  const int half8 = (lane >> 4) << 3;            // 0 or 8
  const int wv    = threadIdx.x >> 5;
  const int bh    = blockIdx.x >> 4;             // 16 query blocks per (b,h)
  const int q0    = (blockIdx.x & 15) * 128 + wv * 16;
  const float scale = 0.125f;                    // 1/sqrt(64)

  const u16* kbase = Kp + (size_t)bh * S_ * DEPTH_;
  const u16* vbase = Vt + (size_t)bh * DEPTH_ * S_;

  // Q tile A-fragments (16 rows x 64 K), loaded once from global
  const u16* qbase = Qp + ((size_t)bh * S_ + (q0 + lm)) * DEPTH_;
  BFrag qf0, qf1;
  qf0.h[0] = *(const u16x8*)(qbase + half8);
  qf0.h[1] = *(const u16x8*)(qbase + half8 + 16);
  qf1.h[0] = *(const u16x8*)(qbase + 32 + half8);
  qf1.h[1] = *(const u16x8*)(qbase + 32 + half8 + 16);

  // --- chunk issue/wait helpers (wave 0 drives the TDM) ---
#if HAVE_TDM
#define ISSUE_K(c) do { if (wv == 0) tdm_issue_2d((unsigned)(((c) & 1) * KBUF_BYTES), \
    kbase + (size_t)(c) * 32 * DEPTH_, DEPTH_, 32, DEPTH_, DEPTH_, 32); } while (0)
#define ISSUE_V(c) do { if (wv == 0) tdm_issue_2d((unsigned)(8192 + ((c) & 1) * VBUF_BYTES), \
    vbase + (size_t)(c) * 32, S_, DEPTH_, S_, 32, DEPTH_); } while (0)
#define WAIT_TDM(n) do { if (wv == 0) __builtin_amdgcn_s_wait_tensorcnt(n); } while (0)
#else
#define ISSUE_K(c) coop_copy_2d(sm.kbuf[(c) & 1], kbase + (size_t)(c) * 32 * DEPTH_, 32, DEPTH_, DEPTH_)
#define ISSUE_V(c) coop_copy_2d(sm.vbuf[(c) & 1], vbase + (size_t)(c) * 32, DEPTH_, 32, S_)
#define WAIT_TDM(n) do { } while (0)
#endif

  float st[8];   // phase A: per-lane running max ; phase B: per-lane exp sums
  float m[8], invl[8];

  // ================= phase A: row max =================
#pragma unroll
  for (int r = 0; r < 8; ++r) st[r] = -3.0e38f;
  ISSUE_K(0);
  for (int c = 0; c < NCH; ++c) {
    if (c + 1 < NCH) { ISSUE_K(c + 1); WAIT_TDM(1); }
    else             { WAIT_TDM(0); }
    __syncthreads();
    const u16* kb = sm.kbuf[c & 1];
#pragma unroll
    for (int t2 = 0; t2 < 2; ++t2) {
      const u16* kr = kb + (t2 * 16 + lm) * DEPTH_;
      BFrag kf0, kf1;
      kf0.h[0] = *(const u16x8*)(kr + half8);
      kf0.h[1] = *(const u16x8*)(kr + half8 + 16);
      kf1.h[0] = *(const u16x8*)(kr + 32 + half8);
      kf1.h[1] = *(const u16x8*)(kr + 32 + half8 + 16);
      v8f s = {};
      s = __builtin_amdgcn_wmma_f32_16x16x32_bf16(false, qf0.v, false, kf0.v, (short)0, s, false, false);
      s = __builtin_amdgcn_wmma_f32_16x16x32_bf16(false, qf1.v, false, kf1.v, (short)0, s, false, false);
#pragma unroll
      for (int r = 0; r < 8; ++r) st[r] = fmaxf(st[r], s[r] * scale);
    }
    __syncthreads();
  }
#pragma unroll
  for (int r = 0; r < 8; ++r) m[r] = rmax16(st[r]);

  // ================= phase B: exp-sum =================
#pragma unroll
  for (int r = 0; r < 8; ++r) st[r] = 0.0f;
  ISSUE_K(0);
  for (int c = 0; c < NCH; ++c) {
    if (c + 1 < NCH) { ISSUE_K(c + 1); WAIT_TDM(1); }
    else             { WAIT_TDM(0); }
    __syncthreads();
    const u16* kb = sm.kbuf[c & 1];
#pragma unroll
    for (int t2 = 0; t2 < 2; ++t2) {
      const u16* kr = kb + (t2 * 16 + lm) * DEPTH_;
      BFrag kf0, kf1;
      kf0.h[0] = *(const u16x8*)(kr + half8);
      kf0.h[1] = *(const u16x8*)(kr + half8 + 16);
      kf1.h[0] = *(const u16x8*)(kr + 32 + half8);
      kf1.h[1] = *(const u16x8*)(kr + 32 + half8 + 16);
      v8f s = {};
      s = __builtin_amdgcn_wmma_f32_16x16x32_bf16(false, qf0.v, false, kf0.v, (short)0, s, false, false);
      s = __builtin_amdgcn_wmma_f32_16x16x32_bf16(false, qf1.v, false, kf1.v, (short)0, s, false, false);
#pragma unroll
      for (int r = 0; r < 8; ++r) st[r] += __expf(s[r] * scale - m[r]);
    }
    __syncthreads();
  }
#pragma unroll
  for (int r = 0; r < 8; ++r) invl[r] = 1.0f / rsum16(st[r]);

  // ===== phase C: normalized attn out + ctx = attn @ V =====
  v8f ctx[4] = {v8f{}, v8f{}, v8f{}, v8f{}};
  float* attnBase = attnOut + ((size_t)bh * S_ + q0) * S_;
  u16*   myLds    = sm.pst + wv * (16 * 32);

  ISSUE_K(0);
  ISSUE_V(0);
  for (int c = 0; c < NCH; ++c) {
    if (c + 1 < NCH) { ISSUE_K(c + 1); ISSUE_V(c + 1); WAIT_TDM(2); }
    else             { WAIT_TDM(0); }
    __syncthreads();
    const u16* kb = sm.kbuf[c & 1];
    const u16* vb = sm.vbuf[c & 1];
#pragma unroll
    for (int t2 = 0; t2 < 2; ++t2) {
      const u16* kr = kb + (t2 * 16 + lm) * DEPTH_;
      BFrag kf0, kf1;
      kf0.h[0] = *(const u16x8*)(kr + half8);
      kf0.h[1] = *(const u16x8*)(kr + half8 + 16);
      kf1.h[0] = *(const u16x8*)(kr + 32 + half8);
      kf1.h[1] = *(const u16x8*)(kr + 32 + half8 + 16);
      v8f s = {};
      s = __builtin_amdgcn_wmma_f32_16x16x32_bf16(false, qf0.v, false, kf0.v, (short)0, s, false, false);
      s = __builtin_amdgcn_wmma_f32_16x16x32_bf16(false, qf1.v, false, kf1.v, (short)0, s, false, false);
#pragma unroll
      for (int r = 0; r < 8; ++r) {
        const int   rr = r + half8;
        const float p  = __expf(s[r] * scale - m[r]) * invl[r];
        attnBase[(size_t)rr * S_ + (c * 32 + t2 * 16 + lm)] = p;  // normalized prob
        myLds[rr * 32 + t2 * 16 + lm] = f2b(p);                   // C-layout -> LDS
      }
    }
    // LDS relayout: 16x32 probability tile -> A-fragment
    BFrag pf;
    pf.h[0] = *(const u16x8*)(myLds + lm * 32 + half8);
    pf.h[1] = *(const u16x8*)(myLds + lm * 32 + half8 + 16);
#pragma unroll
    for (int t = 0; t < 4; ++t) {
      const u16* vr = vb + (t * 16 + lm) * 32 + half8;
      BFrag vf;
      vf.h[0] = *(const u16x8*)(vr);
      vf.h[1] = *(const u16x8*)(vr + 16);
      ctx[t] = __builtin_amdgcn_wmma_f32_16x16x32_bf16(
          false, pf.v, false, vf.v, (short)0, ctx[t], false, false);
    }
    __syncthreads();
  }

  // store ctx merged-heads bf16 (b, s, h*64+dd)
  const int bb = bh >> 4, hh = bh & 15;
#pragma unroll
  for (int t = 0; t < 4; ++t)
#pragma unroll
    for (int r = 0; r < 8; ++r) {
      const int row = q0 + r + half8;
      const int col = hh * DEPTH_ + t * 16 + lm;
      ctxB[((size_t)bb * S_ + row) * D_ + col] = f2b(ctx[t][r]);
    }
}

extern "C" void kernel_launch(void* const* d_in, const int* in_sizes, int n_in,
                              void* d_out, int out_size, void* d_ws, size_t ws_size,
                              hipStream_t stream) {
  (void)in_sizes; (void)n_in; (void)out_size; (void)ws_size;
  const float* q       = (const float*)d_in[0];
  const float* k       = (const float*)d_in[1];
  const float* v       = (const float*)d_in[2];
  const float* wq_w    = (const float*)d_in[3];
  const float* wq_b    = (const float*)d_in[4];
  const float* wk_w    = (const float*)d_in[5];
  const float* wk_b    = (const float*)d_in[6];
  const float* wv_w    = (const float*)d_in[7];
  const float* wv_b    = (const float*)d_in[8];
  const float* dense_w = (const float*)d_in[9];
  const float* dense_b = (const float*)d_in[10];

  float* out  = (float*)d_out;                                 // B*S*D
  float* attn = out + (size_t)B_ * S_ * D_;                    // B*H*S*S

  const size_t MD = (size_t)B_ * S_ * D_;                      // 4,194,304
  const size_t WD = (size_t)D_ * D_;                           // 1,048,576
  u16* ws   = (u16*)d_ws;
  u16* qb   = ws;
  u16* kb   = qb + MD;
  u16* vb   = kb + MD;
  u16* wqb  = vb + MD;
  u16* wkb  = wqb + WD;
  u16* wvb  = wkb + WD;
  u16* wdb  = wvb + WD;
  u16* Qp   = wdb + WD;
  u16* Kp   = Qp + MD;
  u16* Vt   = Kp + MD;
  u16* ctxb = Vt + MD;

  // fp32 -> bf16 conversions
  cvt_bf16_kernel<<<1024, 256, 0, stream>>>(q, qb, (int)MD);
  cvt_bf16_kernel<<<1024, 256, 0, stream>>>(k, kb, (int)MD);
  cvt_bf16_kernel<<<1024, 256, 0, stream>>>(v, vb, (int)MD);
  cvt_bf16_kernel<<<512, 256, 0, stream>>>(wq_w, wqb, (int)WD);
  cvt_bf16_kernel<<<512, 256, 0, stream>>>(wk_w, wkb, (int)WD);
  cvt_bf16_kernel<<<512, 256, 0, stream>>>(wv_w, wvb, (int)WD);
  cvt_bf16_kernel<<<512, 256, 0, stream>>>(dense_w, wdb, (int)WD);

  // QKV projections: M=4096, N=1024, K=1024 -> 4096 waves -> 512 blocks of 8 waves
  const int M = B_ * S_, N = D_, K = D_;
  const int gemmBlocks = (M / 16) * (N / 64) / 8;
  gemm_xwt_kernel<<<gemmBlocks, 256, 0, stream>>>(qb, wqb, wq_b, Qp, nullptr, M, N, K, 0);
  gemm_xwt_kernel<<<gemmBlocks, 256, 0, stream>>>(kb, wkb, wk_b, Kp, nullptr, M, N, K, 0);
  gemm_xwt_kernel<<<gemmBlocks, 256, 0, stream>>>(vb, wvb, wv_b, Vt, nullptr, M, N, K, 1);

  // fused attention: 32 (b,h) x 16 query blocks of 128 rows
  attn_kernel<<<B_ * H_ * (S_ / 128), 256, 0, stream>>>(Qp, Kp, Vt, attn, ctxb);

  // output projection fp32
  gemm_xwt_kernel<<<gemmBlocks, 256, 0, stream>>>(ctxb, wdb, dense_b, nullptr, out, M, N, K, 2);
}
